// CrossAttention_59055800320135
// MI455X (gfx1250) — matrix operand order repaired and checked by reading, not compiled, not run on hardware
//
#include <hip/hip_runtime.h>
#include <hip/hip_bf16.h>

// Problem constants (from reference): B=4, N=4096, C=1024, H=16, D=64
#define BB   4
#define NSEQ 4096
#define CDIM 1024
#define NH   16
#define HD   64
#define ATT_SCALE 0.125f  // 64^-0.5

typedef __attribute__((ext_vector_type(16))) __bf16 v16bf;
typedef __attribute__((ext_vector_type(8)))  __bf16 v8bf;
typedef __attribute__((ext_vector_type(8)))  float  v8f;
typedef unsigned short ushort_t;

static __device__ __forceinline__ __bf16 f2bf(float f) {
    return (__bf16)f;                       // native v_cvt (RNE) on gfx1250
}
static __device__ __forceinline__ ushort_t f2bfu(float f) {
    __bf16 h = (__bf16)f;
    return __builtin_bit_cast(ushort_t, h);
}
static __device__ __forceinline__ float bfu2f(ushort_t h) {
    unsigned u = ((unsigned)h) << 16;
    return __builtin_bit_cast(float, u);
}
static __device__ __forceinline__ unsigned pack_bf2(float lo, float hi) {
    return (unsigned)f2bfu(lo) | ((unsigned)f2bfu(hi) << 16);
}

// Build a v16bf fragment from two 16B chunks
static __device__ __forceinline__ v16bf frag_cat(v8bf lo, v8bf hi) {
    v16bf r;
    for (int i = 0; i < 8; ++i) { r[i] = lo[i]; r[8 + i] = hi[i]; }
    return r;
}

// ---------------------------------------------------------------------------
// fp32 -> bf16 bulk convert (8 elements / thread, 16B stores)
// ---------------------------------------------------------------------------
__global__ __launch_bounds__(256) void cvt_kernel(
    const float* __restrict__ x, ushort_t* __restrict__ y)
{
    const size_t idx = ((size_t)blockIdx.x * 256 + threadIdx.x) * 8;
    const float4* p = (const float4*)(x + idx);
    float4 a = p[0], b = p[1];
    v8bf o;
    o[0] = f2bf(a.x); o[1] = f2bf(a.y); o[2] = f2bf(a.z); o[3] = f2bf(a.w);
    o[4] = f2bf(b.x); o[5] = f2bf(b.y); o[6] = f2bf(b.z); o[7] = f2bf(b.w);
    *(v8bf*)(y + idx) = o;
}

// ---------------------------------------------------------------------------
// GEMM: Y[M,N] = act(A[M,K] @ W[K,N] + bias);  A bf16, W/bias fp32, Y bf16
//   Block tile 256x64, 8 waves; wave = 32 rows (2 m-tiles) x 64 cols (4 n-tiles)
//   = 8 WMMA per 32-K chunk.  W chunk [32k x 64n] double-buffered in LDS
//   (bf16, [n][k] so k is contiguous per lane) -> one barrier per chunk.
// ---------------------------------------------------------------------------
template<int ACT>   // ACT: 0 none, 1 relu, 2 sigmoid
__global__ __launch_bounds__(256) void gemm_kernel(
    const ushort_t* __restrict__ A, const float* __restrict__ W,
    const float* __restrict__ bias, ushort_t* __restrict__ Y,
    int M, int K, int Nn)
{
    __shared__ __bf16 BT[2][64][40];   // [buf][n][k], 80B row stride

    const int t    = threadIdx.x;
    const int lane = t & 31, wave = t >> 5;
    const int lm   = lane & 15, kh = lane >> 4;
    const int m0   = blockIdx.x * 256;
    const int n0   = blockIdx.y * 64;

    // staging coords: each thread covers 2(k) x 4(n)
    const int sk = (t >> 4) * 2;      // 0,2,...,30
    const int sn = (t & 15) * 4;      // 0,4,...,60

    // compute coords
    const int mrow = m0 + wave * 32 + lm;   // m-tile 0; m-tile 1 = +16

    v8f acc[8];
    for (int q = 0; q < 8; ++q)
        for (int j = 0; j < 8; ++j) acc[q][j] = 0.f;

    auto stage = [&](int k0, int p) {
        const float* wr0 = W + (size_t)(k0 + sk) * Nn + n0 + sn;
        float4 r0 = *(const float4*)wr0;
        float4 r1 = *(const float4*)(wr0 + Nn);
        *(unsigned*)&BT[p][sn + 0][sk] = pack_bf2(r0.x, r1.x);
        *(unsigned*)&BT[p][sn + 1][sk] = pack_bf2(r0.y, r1.y);
        *(unsigned*)&BT[p][sn + 2][sk] = pack_bf2(r0.z, r1.z);
        *(unsigned*)&BT[p][sn + 3][sk] = pack_bf2(r0.w, r1.w);
    };

    stage(0, 0);
    __syncthreads();

    int p = 0;
    for (int k0 = 0; k0 < K; k0 += 32) {
        if (k0 + 32 < K) stage(k0 + 32, p ^ 1);   // prefetch next chunk

        // A fragments: lane = row, k = kb..kb+7 and kb+16..kb+23 (kb = k0+kh*8)
        const ushort_t* Ap0 = A + (size_t)mrow * K + k0 + kh * 8;
        const ushort_t* Ap1 = Ap0 + (size_t)16 * K;
        v16bf a0 = frag_cat(*(const v8bf*)Ap0, *(const v8bf*)(Ap0 + 16));
        v16bf a1 = frag_cat(*(const v8bf*)Ap1, *(const v8bf*)(Ap1 + 16));

        for (int nt = 0; nt < 4; ++nt) {
            const int n = nt * 16 + lm;
            // B fragment: lane = col n, k = kh*8..kh*8+7 and kh*8+16..kh*8+23
            v16bf bfr = frag_cat(*(const v8bf*)&BT[p][n][kh * 8],
                                 *(const v8bf*)&BT[p][n][kh * 8 + 16]);
            acc[nt] = __builtin_amdgcn_wmma_f32_16x16x32_bf16(
                false, a0, false, bfr, (short)0, acc[nt], false, false);
            acc[4 + nt] = __builtin_amdgcn_wmma_f32_16x16x32_bf16(
                false, a1, false, bfr, (short)0, acc[4 + nt], false, false);
        }
        __syncthreads();   // readers done with buf p, writers done with buf p^1
        p ^= 1;
    }

    // epilogue: row = m0 + wave*32 + mt*16 + kh*8 + j, col = n0 + nt*16 + lm
    for (int mt = 0; mt < 2; ++mt) {
        const int row0 = m0 + wave * 32 + mt * 16 + kh * 8;
        for (int nt = 0; nt < 4; ++nt) {
            const int c = n0 + nt * 16 + lm;
            const float bv = bias ? bias[c] : 0.f;
            for (int j = 0; j < 8; ++j) {
                float v = acc[mt * 4 + nt][j] + bv;
                if (ACT == 1) v = fmaxf(v, 0.f);
                if (ACT == 2) v = 1.0f / (1.0f + __expf(-v));
                Y[(size_t)(row0 + j) * Nn + c] = f2bfu(v);
            }
        }
    }
}

// ---------------------------------------------------------------------------
// Split kv into kT [B,H,D,N] and (v*gate)^T [B,H,E,N] (bf16)
// ---------------------------------------------------------------------------
__global__ __launch_bounds__(256) void rearrange_kernel(
    const ushort_t* __restrict__ kv, const ushort_t* __restrict__ gate,
    ushort_t* __restrict__ kT, ushort_t* __restrict__ vgT)
{
    const size_t idx = (size_t)blockIdx.x * 256 + threadIdx.x; // [b][h][n][d]
    const int d = (int)(idx & 63);
    const int n = (int)((idx >> 6) & 4095);
    const int h = (int)((idx >> 18) & 15);
    const int b = (int)(idx >> 22);
    const size_t tok = (size_t)b * NSEQ + n;
    const ushort_t kk = kv[tok * 2048 + h * 64 + d];
    const ushort_t vv = kv[tok * 2048 + 1024 + h * 64 + d];
    const ushort_t gg = gate[tok * 1024 + h * 64 + d];
    const size_t bh = (size_t)b * NH + h;
    const size_t off = (bh * 64 + d) * NSEQ + n;
    kT[off]  = kk;
    vgT[off] = f2bfu(bfu2f(vv) * bfu2f(gg));
}

// ---------------------------------------------------------------------------
// ctx[b,h,d,e] = softmax_d( SCALE * sum_n kT[d,n] * vgT[e,n] )
// One WG per (b,h). 8 waves: wave -> (mt = w&3, n-pair = w>>2), K=4096.
// A and B fragments are both contiguous 16B loads from [64][4096] rows.
// ---------------------------------------------------------------------------
__global__ __launch_bounds__(256) void ctx_kernel(
    const ushort_t* __restrict__ kT, const ushort_t* __restrict__ vgT,
    float* __restrict__ ctx)
{
    __shared__ float tile[64][65];
    const int bh = blockIdx.x;
    const int t = threadIdx.x, lane = t & 31, wave = t >> 5;
    const int lm = lane & 15, kh = lane >> 4;
    const int mt = wave & 3, np = wave >> 2;

    const ushort_t* Abase = kT  + (size_t)bh * 64 * NSEQ;
    const ushort_t* Bbase = vgT + (size_t)bh * 64 * NSEQ;

    v8f acc[2];
    for (int q = 0; q < 2; ++q)
        for (int j = 0; j < 8; ++j) acc[q][j] = 0.f;

    const ushort_t* Arow  = Abase + (size_t)(mt * 16 + lm) * NSEQ;
    const ushort_t* Brow0 = Bbase + (size_t)((np * 2 + 0) * 16 + lm) * NSEQ;
    const ushort_t* Brow1 = Bbase + (size_t)((np * 2 + 1) * 16 + lm) * NSEQ;

    for (int k0 = 0; k0 < NSEQ; k0 += 32) {
        const ushort_t* ap = Arow + k0 + kh * 16;
        v16bf a = frag_cat(*(const v8bf*)ap, *(const v8bf*)(ap + 8));

        const ushort_t* bp0 = Brow0 + k0 + kh * 16;
        v16bf b0 = frag_cat(*(const v8bf*)bp0, *(const v8bf*)(bp0 + 8));
        acc[0] = __builtin_amdgcn_wmma_f32_16x16x32_bf16(
            false, a, false, b0, (short)0, acc[0], false, false);

        const ushort_t* bp1 = Brow1 + k0 + kh * 16;
        v16bf b1 = frag_cat(*(const v8bf*)bp1, *(const v8bf*)(bp1 + 8));
        acc[1] = __builtin_amdgcn_wmma_f32_16x16x32_bf16(
            false, a, false, b1, (short)0, acc[1], false, false);
    }

    for (int q = 0; q < 2; ++q) {
        const int nt = np * 2 + q;
        for (int j = 0; j < 8; ++j)
            tile[mt * 16 + kh * 8 + j][nt * 16 + lm] = acc[q][j] * ATT_SCALE;
    }
    __syncthreads();

    if (t < 64) {                       // softmax over d (rows) per column e
        const int e = t;
        float mx = tile[0][e];
        for (int d = 1; d < 64; ++d) mx = fmaxf(mx, tile[d][e]);
        float s = 0.f;
        for (int d = 0; d < 64; ++d) s += __expf(tile[d][e] - mx);
        const float inv = 1.0f / s;
        float* cp = ctx + (size_t)bh * 4096;
        for (int d = 0; d < 64; ++d)
            cp[d * 64 + e] = __expf(tile[d][e] - mx) * inv;
    }
}

// ---------------------------------------------------------------------------
// o[b,n,h*64+e] = sum_d xbf[b,n,h*64+d] * ctx[b,h,d,e]
// grid.x = M tiles (4096/128), grid.y = b*H+h. ctx staged to LDS (bf16, e-major).
// ---------------------------------------------------------------------------
__global__ __launch_bounds__(256) void out_kernel(
    const ushort_t* __restrict__ xbf, const float* __restrict__ ctx,
    float* __restrict__ o)
{
    __shared__ __bf16 ctxT[64][72];   // [e][d], 144B row stride (16B multiple)
    const int t = threadIdx.x, lane = t & 31, wave = t >> 5;
    const int lm = lane & 15, kh = lane >> 4;
    const int bh = blockIdx.y;
    const int b = bh >> 4, h = bh & 15;
    const int m0 = blockIdx.x * 128;

    {   // cooperative ctx -> LDS transposed bf16
        const int d = t >> 2, e0 = (t & 3) * 16;
        const float* cp = ctx + (size_t)bh * 4096 + d * 64 + e0;
        for (int j = 0; j < 16; ++j) ctxT[e0 + j][d] = f2bf(cp[j]);
    }
    __syncthreads();

    const int nrow = m0 + wave * 16 + lm;
    const ushort_t* xrow = xbf + ((size_t)b * NSEQ + nrow) * CDIM + h * 64;

    v8f acc[4];
    for (int nt = 0; nt < 4; ++nt)
        for (int j = 0; j < 8; ++j) acc[nt][j] = 0.f;

    for (int kt = 0; kt < 2; ++kt) {
        const ushort_t* ap = xrow + kt * 32 + kh * 16;
        v16bf a = frag_cat(*(const v8bf*)ap, *(const v8bf*)(ap + 8));
        for (int nt = 0; nt < 4; ++nt) {
            const int e = nt * 16 + lm;
            const int kb = kt * 32 + kh * 16;
            v16bf bfr = frag_cat(*(const v8bf*)&ctxT[e][kb],
                                 *(const v8bf*)&ctxT[e][kb + 8]);
            acc[nt] = __builtin_amdgcn_wmma_f32_16x16x32_bf16(
                false, a, false, bfr, (short)0, acc[nt], false, false);
        }
    }

    const int row0 = m0 + wave * 16 + kh * 8;
    for (int nt = 0; nt < 4; ++nt) {
        const int e = nt * 16 + lm;
        for (int j = 0; j < 8; ++j)
            o[((size_t)b * NSEQ + row0 + j) * CDIM + h * 64 + e] = acc[nt][j];
    }
}

// ---------------------------------------------------------------------------
extern "C" void kernel_launch(void* const* d_in, const int* in_sizes, int n_in,
                              void* d_out, int out_size, void* d_ws, size_t ws_size,
                              hipStream_t stream)
{
    (void)in_sizes; (void)n_in; (void)out_size; (void)ws_size;
    const float* xs[2]  = { (const float*)d_in[0], (const float*)d_in[1] };
    const float* Wkv[2] = { (const float*)d_in[2], (const float*)d_in[3] };
    const float* gw1[2] = { (const float*)d_in[4], (const float*)d_in[8] };
    const float* gb1[2] = { (const float*)d_in[5], (const float*)d_in[9] };
    const float* gw2[2] = { (const float*)d_in[6], (const float*)d_in[10] };
    const float* gb2[2] = { (const float*)d_in[7], (const float*)d_in[11] };

    const size_t M = (size_t)BB * NSEQ;  // 16384
    char* ws = (char*)d_ws;
    ushort_t* xbf[2];
    xbf[0] = (ushort_t*)ws;  ws += M * CDIM * sizeof(ushort_t);
    xbf[1] = (ushort_t*)ws;  ws += M * CDIM * sizeof(ushort_t);
    ushort_t* kv_buf = (ushort_t*)ws;  ws += M * 2048 * sizeof(ushort_t);
    ushort_t* h_buf  = (ushort_t*)ws;  ws += M * 1024 * sizeof(ushort_t);
    ushort_t* g_buf  = (ushort_t*)ws;  ws += M * 1024 * sizeof(ushort_t);
    ushort_t* kT_buf = (ushort_t*)ws;  ws += M * 1024 * sizeof(ushort_t);
    ushort_t* vg_buf = (ushort_t*)ws;  ws += M * 1024 * sizeof(ushort_t);
    float* ctxb[2];
    ctxb[0] = (float*)ws;  ws += (size_t)BB * NH * 64 * 64 * sizeof(float);
    ctxb[1] = (float*)ws;

    const dim3 blk(256);
    const int cvt_blocks = (int)(M * CDIM / (256 * 8));   // 8192
    cvt_kernel<<<dim3(cvt_blocks), blk, 0, stream>>>(xs[0], xbf[0]);
    cvt_kernel<<<dim3(cvt_blocks), blk, 0, stream>>>(xs[1], xbf[1]);

    for (int s = 0; s < 2; ++s) {
        gemm_kernel<0><<<dim3(64, 32), blk, 0, stream>>>(
            xbf[s], Wkv[s], nullptr, kv_buf, 16384, 1024, 2048);
        gemm_kernel<1><<<dim3(64, 16), blk, 0, stream>>>(
            xbf[s], gw1[s], gb1[s], h_buf, 16384, 1024, 1024);
        gemm_kernel<2><<<dim3(64, 16), blk, 0, stream>>>(
            h_buf, gw2[s], gb2[s], g_buf, 16384, 1024, 1024);
        rearrange_kernel<<<dim3(65536), blk, 0, stream>>>(kv_buf, g_buf, kT_buf, vg_buf);
        ctx_kernel<<<dim3(64), blk, 0, stream>>>(kT_buf, vg_buf, ctxb[s]);
    }
    float* o1 = (float*)d_out;
    float* o2 = o1 + (size_t)BB * NSEQ * CDIM;
    out_kernel<<<dim3(32, 64), blk, 0, stream>>>(xbf[0], ctxb[1], o1);  // x1 uses ctx2
    out_kernel<<<dim3(32, 64), blk, 0, stream>>>(xbf[1], ctxb[0], o2);  // x2 uses ctx1
}
